// ChEBIRecNN_76416058130807
// MI455X (gfx1250) — matrix operand order repaired and checked by reading, not compiled, not run on hardware
//
#include <hip/hip_runtime.h>

// ---------------------------------------------------------------------------
// Problem constants (ChEBIRecNN): D=256 DAGs, N=512 nodes, ATOM=62, L=200,
// F=8 carry slots, C=500 classes.  M = D*N = 131072 rows for stage 1.
// ---------------------------------------------------------------------------
#define NDAG   256
#define NNODE  512
#define NATOM  62
#define LDIM   200
#define FWIN   8
#define NCLS   500
#define MROWS  (NDAG * NNODE)

// Padded sizes for WMMA (K multiple of 32, N multiple of 16)
#define KP1    64     // 62  -> 64   (W1)
#define KP2    224    // 200 -> 224  (W2, Wm2)
#define KPM1   416    // 400 -> 416  (Wm1)
#define NP     208    // 200 -> 208

typedef __attribute__((ext_vector_type(16))) __bf16          v16bf;
typedef __attribute__((ext_vector_type(16))) unsigned short  v16u;
typedef __attribute__((ext_vector_type(8)))  unsigned short  v8u;
typedef __attribute__((ext_vector_type(8)))  float           v8f;

__device__ __forceinline__ unsigned short f32_to_bf16(float f) {
  union { float f; unsigned u; } c; c.f = f;
  unsigned r = c.u + 0x7FFFu + ((c.u >> 16) & 1u);   // round-to-nearest-even
  return (unsigned short)(r >> 16);
}
__device__ __forceinline__ float bf16_to_f32(unsigned short h) {
  union { unsigned u; float f; } c; c.u = ((unsigned)h) << 16;
  return c.f;
}

// ---------------------------------------------------------------------------
// One 16x16 output tile: A (bf16, row-major M x Kp, in LDS, fully zero-padded
// to Kp) x Bt (bf16, TRANSPOSED N x Kp, in global).  K steps of 32.
//
// Fragment layouts per CDNA5 ISA 7.12.2 give contiguous chunks:
//   A: lane L, elem i -> K = 32kk + 8*hi + i (i<8) | 32kk + 16 + 8*hi + (i-8)
//      => two 16B ds_load_b128 per k-step.
//   B: lane L (col = L%16), elem i -> K = 32kk + 8*hi + i (i<8) | 32kk + 16 + 8*hi + (i-8)
//      => two 16B global_load_b128 per k-step.
//   C: VGPR v -> row M = v + 8*hi, col N = lane%16
// ---------------------------------------------------------------------------
template <int KSTEPS>
__device__ __forceinline__ v8f wmma_rowtile(const unsigned short* A, int lda,
                                            const unsigned short* Bt, int ldbt,
                                            int n0, int lane, v8f c) {
  const int hi  = (lane & 16) ? 1 : 0;
  const int col = lane & 15;
  const unsigned short* arow = A + (size_t)(lane & 15) * lda;
  const unsigned short* brow = Bt + (size_t)(n0 + col) * ldbt;
  __builtin_prefetch(brow, 0, 0);
#pragma unroll
  for (int kk = 0; kk < KSTEPS; ++kk) {
    v8u a0 = *(const v8u*)(arow + kk * 32 + hi * 8);
    v8u a1 = *(const v8u*)(arow + kk * 32 + 16 + hi * 8);
    v8u b0 = *(const v8u*)(brow + kk * 32 + hi * 8);
    v8u b1 = *(const v8u*)(brow + kk * 32 + 16 + hi * 8);
    v16u au = __builtin_shufflevector(a0, a1, 0,1,2,3,4,5,6,7,8,9,10,11,12,13,14,15);
    v16u bu = __builtin_shufflevector(b0, b1, 0,1,2,3,4,5,6,7,8,9,10,11,12,13,14,15);
    c = __builtin_amdgcn_wmma_f32_16x16x32_bf16(
        false, __builtin_bit_cast(v16bf, au),
        false, __builtin_bit_cast(v16bf, bu),
        (short)0, c, false, false);
  }
  return c;
}

// ---------------------------------------------------------------------------
// Pack fp32 weight (K x N) -> zero-padded TRANSPOSED bf16 (Np x Kp).
// ---------------------------------------------------------------------------
__global__ void pack_bf16_t(const float* __restrict__ src, unsigned short* __restrict__ dst,
                            int K, int N, int Kp, int Np) {
  int idx = blockIdx.x * blockDim.x + threadIdx.x;
  if (idx >= Kp * Np) return;
  int n = idx / Kp, k = idx % Kp;
  float v = (k < K && n < N) ? src[k * N + n] : 0.f;
  dst[idx] = f32_to_bf16(v);
}

// ---------------------------------------------------------------------------
// Stage 1: a = relu(relu(X@W1+b1)@W2+b2),  X: (131072 x 62) fp32.
// 64-row tile / workgroup, 8 waves.  Output a as bf16 (131072 x 200).
// ---------------------------------------------------------------------------
#define MT 64
__global__ __launch_bounds__(256) void stage1_kernel(
    const float* __restrict__ X,
    const unsigned short* __restrict__ W1t, const float* __restrict__ b1,
    const unsigned short* __restrict__ W2t, const float* __restrict__ b2,
    unsigned short* __restrict__ a_bf) {
  __shared__ __align__(16) unsigned short Xs[MT][KP1];   // 8 KB
  __shared__ __align__(16) unsigned short H1[MT][KP2];   // 28 KB (K-padded to 224)
  const int tid = threadIdx.x;
  const int m0  = blockIdx.x * MT;

  // Stage X tile into LDS as bf16 (zero-pad K 62->64); zero H1 pad columns.
  for (int i = tid; i < MT * KP1; i += 256) {
    int r = i / KP1, k = i % KP1;
    float v = (k < NATOM) ? X[(size_t)(m0 + r) * NATOM + k] : 0.f;
    Xs[r][k] = f32_to_bf16(v);
  }
  for (int i = tid; i < MT * (KP2 - NP); i += 256)
    H1[i / (KP2 - NP)][NP + i % (KP2 - NP)] = 0;
  __syncthreads();

  const int lane = tid & 31, wave = tid >> 5;
  const bool hi  = (lane & 16) != 0;
  const int col  = lane & 15;

  // GEMM1: (64x64) @ (64x208) -> H1, fused bias+relu, bf16 into LDS
  for (int t = wave; t < 4 * 13; t += 8) {
    int ms = t / 13, nt = t % 13;
    v8f c = {};
    c = wmma_rowtile<KP1 / 32>(&Xs[ms * 16][0], KP1, W1t, KP1, nt * 16, lane, c);
    int n = nt * 16 + col;
    float bias = (n < LDIM) ? b1[n] : 0.f;
#pragma unroll
    for (int v = 0; v < 8; ++v) {
      float r = c[v] + bias;
      r = (n < LDIM && r > 0.f) ? r : 0.f;
      H1[ms * 16 + v + (hi ? 8 : 0)][n] = f32_to_bf16(r);
    }
  }
  __syncthreads();

  // GEMM2: (64x224) @ (224x208, K zero-padded) -> a, fused bias+relu, bf16 out
  for (int t = wave; t < 4 * 13; t += 8) {
    int ms = t / 13, nt = t % 13;
    v8f c = {};
    c = wmma_rowtile<KP2 / 32>(&H1[ms * 16][0], KP2, W2t, KP2, nt * 16, lane, c);
    int n = nt * 16 + col;
    if (n < LDIM) {
      float bias = b2[n];
#pragma unroll
      for (int v = 0; v < 8; ++v) {
        float r = fmaxf(c[v] + bias, 0.f);
        int m = m0 + ms * 16 + v + (hi ? 8 : 0);
        a_bf[(size_t)m * LDIM + n] = f32_to_bf16(r);
      }
    }
  }
}

// ---------------------------------------------------------------------------
// Stage 2: recursive scan.  16 DAGs per workgroup, 16 workgroups, 8 waves.
// carry ring (16 x 8 x 208 fp32) lives in LDS for all 504 steps.
// Per step: attention (VALU fp32) -> inp bf16 -> WMMA (16x416)@(416x208)
// -> relu -> WMMA (16x224)@(224x208) -> relu + prev -> carry slot s%8.
// ---------------------------------------------------------------------------
#define DAGS 16
__global__ __launch_bounds__(256) void stage2_kernel(
    const unsigned short* __restrict__ a_bf,
    const unsigned short* __restrict__ Wm1t, const float* __restrict__ bm1,
    const unsigned short* __restrict__ Wm2t, const float* __restrict__ bm2,
    const float* __restrict__ attn_w, float* __restrict__ lasts) {
  __shared__ float                       carry[DAGS][FWIN][NP];  // 106.5 KB
  __shared__ float                       prev [DAGS][NP];        // 13.3 KB
  __shared__ __align__(16) unsigned short inp [DAGS][KPM1];      // 13.3 KB
  __shared__ __align__(16) unsigned short hmid[DAGS][KP2];       // 7 KB
  __shared__ float                       sc   [DAGS][FWIN];      // 0.5 KB

  const int tid  = threadIdx.x;
  const int d0   = blockIdx.x * DAGS;
  const int lane = tid & 31, wave = tid >> 5;
  const bool hi  = (lane & 16) != 0;
  const int col  = lane & 15;

  // init carry ring from a[d, 0..7] (bf16 -> fp32), zero pads
  for (int i = tid; i < DAGS * FWIN * NP; i += 256) {
    int g = i / (FWIN * NP), r = (i / NP) % FWIN, j = i % NP;
    float v = 0.f;
    if (j < LDIM)
      v = bf16_to_f32(a_bf[((size_t)(d0 + g) * NNODE + r) * LDIM + j]);
    carry[g][r][j] = v;
  }
  // static zero pads of inp / hmid (rewritten regions are [0,400) / [0,208))
  for (int i = tid; i < DAGS * 16; i += 256)
    inp[i / 16][400 + (i % 16)] = 0;
  for (int i = tid; i < DAGS * KP2; i += 256)
    hmid[i / KP2][i % KP2] = 0;
  __syncthreads();

  for (int s = 0; s < NNODE - FWIN; ++s) {
    // ---- attention scores: 128 dot products of length 200 ----
    for (int i = tid; i < DAGS * FWIN; i += 256) {
      int g = i / FWIN, r = i % FWIN;
      float acc = 0.f;
      for (int j = 0; j < LDIM; ++j) acc += carry[g][r][j] * attn_w[j];
      sc[g][r] = acc;
    }
    __syncthreads();
    // ---- softmax over 8 slots (one thread per DAG) ----
    if (tid < DAGS) {
      float m = sc[tid][0];
#pragma unroll
      for (int r = 1; r < FWIN; ++r) m = fmaxf(m, sc[tid][r]);
      float sum = 0.f;
#pragma unroll
      for (int r = 0; r < FWIN; ++r) { float e = expf(sc[tid][r] - m); sc[tid][r] = e; sum += e; }
      float inv = 1.f / sum;
#pragma unroll
      for (int r = 0; r < FWIN; ++r) sc[tid][r] *= inv;
    }
    __syncthreads();
    // ---- prev = sum_r sc*carry ; build inp = [prev_bf16 | at_bf16] ----
    for (int i = tid; i < DAGS * NP; i += 256) {
      int g = i / NP, j = i % NP;
      float p = 0.f;
#pragma unroll
      for (int r = 0; r < FWIN; ++r) p += sc[g][r] * carry[g][r][j];
      prev[g][j] = p;
      if (j < LDIM) {
        inp[g][j] = f32_to_bf16(p);
        inp[g][LDIM + j] = a_bf[((size_t)(d0 + g) * NNODE + FWIN + s) * LDIM + j];
      }
    }
    __syncthreads();
    // ---- GEMM1: (16 x 416) @ (416 x 208) + bm1, relu -> hmid bf16 ----
    for (int nt = wave; nt < 13; nt += 8) {
      v8f c = {};
      c = wmma_rowtile<KPM1 / 32>(&inp[0][0], KPM1, Wm1t, KPM1, nt * 16, lane, c);
      int n = nt * 16 + col;
      float bias = (n < LDIM) ? bm1[n] : 0.f;
#pragma unroll
      for (int v = 0; v < 8; ++v) {
        float r = c[v] + bias;
        r = (n < LDIM && r > 0.f) ? r : 0.f;
        hmid[v + (hi ? 8 : 0)][n] = f32_to_bf16(r);
      }
    }
    __syncthreads();
    // ---- GEMM2: (16 x 224) @ (224 x 208) + bm2, relu, + prev -> carry ----
    const int slot = s & (FWIN - 1);
    for (int nt = wave; nt < 13; nt += 8) {
      v8f c = {};
      c = wmma_rowtile<KP2 / 32>(&hmid[0][0], KP2, Wm2t, KP2, nt * 16, lane, c);
      int n = nt * 16 + col;
      float bias = (n < LDIM) ? bm2[n] : 0.f;
#pragma unroll
      for (int v = 0; v < 8; ++v) {
        int g = v + (hi ? 8 : 0);
        float r = c[v] + bias;
        r = (n < LDIM && r > 0.f) ? r : 0.f;
        carry[g][slot][n] = r + prev[g][n];
      }
    }
    __syncthreads();
  }

  // last written slot is (503 % 8) == 7
  for (int i = tid; i < DAGS * LDIM; i += 256) {
    int g = i / LDIM, j = i % LDIM;
    lasts[(size_t)(d0 + g) * LDIM + j] = carry[g][(NNODE - FWIN - 1) & (FWIN - 1)][j];
  }
}

// ---------------------------------------------------------------------------
// Stage 3: pooled = attention(dag_w, lasts); out = relu(relu(pooled@Wf1+bf1)
// @Wf2+bf2)@Wf3+bf3.  Tiny: one block, fp32 VALU.
// ---------------------------------------------------------------------------
__global__ __launch_bounds__(256) void final_kernel(
    const float* __restrict__ lasts, const float* __restrict__ dag_w,
    const float* __restrict__ Wf1, const float* __restrict__ bf1,
    const float* __restrict__ Wf2, const float* __restrict__ bf2,
    const float* __restrict__ Wf3, const float* __restrict__ bf3,
    float* __restrict__ out) {
  __shared__ float sc[NDAG];
  __shared__ float pooled[LDIM], t1[LDIM], t2[LDIM];
  const int tid = threadIdx.x;

  float acc = 0.f;
  for (int j = 0; j < LDIM; ++j) acc += lasts[(size_t)tid * LDIM + j] * dag_w[j];
  sc[tid] = acc;
  __syncthreads();
  if (tid == 0) {
    float m = sc[0];
    for (int d = 1; d < NDAG; ++d) m = fmaxf(m, sc[d]);
    float sum = 0.f;
    for (int d = 0; d < NDAG; ++d) { float e = expf(sc[d] - m); sc[d] = e; sum += e; }
    float inv = 1.f / sum;
    for (int d = 0; d < NDAG; ++d) sc[d] *= inv;
  }
  __syncthreads();
  if (tid < LDIM) {
    float p = 0.f;
    for (int d = 0; d < NDAG; ++d) p += sc[d] * lasts[(size_t)d * LDIM + tid];
    pooled[tid] = p;
  }
  __syncthreads();
  if (tid < LDIM) {
    float a = bf1[tid];
    for (int k = 0; k < LDIM; ++k) a += pooled[k] * Wf1[k * LDIM + tid];
    t1[tid] = fmaxf(a, 0.f);
  }
  __syncthreads();
  if (tid < LDIM) {
    float a = bf2[tid];
    for (int k = 0; k < LDIM; ++k) a += t1[k] * Wf2[k * LDIM + tid];
    t2[tid] = fmaxf(a, 0.f);
  }
  __syncthreads();
  for (int n = tid; n < NCLS; n += 256) {
    float a = bf3[n];
    for (int k = 0; k < LDIM; ++k) a += t2[k] * Wf3[k * NCLS + n];
    out[n] = a;
  }
}

// ---------------------------------------------------------------------------
extern "C" void kernel_launch(void* const* d_in, const int* in_sizes, int n_in,
                              void* d_out, int out_size, void* d_ws, size_t ws_size,
                              hipStream_t stream) {
  (void)in_sizes; (void)n_in; (void)out_size; (void)ws_size;
  const float* X      = (const float*)d_in[0];
  const float* W1     = (const float*)d_in[1];
  const float* b1     = (const float*)d_in[2];
  const float* W2     = (const float*)d_in[3];
  const float* b2     = (const float*)d_in[4];
  const float* Wm1    = (const float*)d_in[5];
  const float* bm1    = (const float*)d_in[6];
  const float* Wm2    = (const float*)d_in[7];
  const float* bm2    = (const float*)d_in[8];
  const float* attn_w = (const float*)d_in[9];
  const float* dag_w  = (const float*)d_in[10];
  const float* Wf1    = (const float*)d_in[11];
  const float* bf1    = (const float*)d_in[12];
  const float* Wf2    = (const float*)d_in[13];
  const float* bf2    = (const float*)d_in[14];
  const float* Wf3    = (const float*)d_in[15];
  const float* bf3    = (const float*)d_in[16];
  float* out = (float*)d_out;

  // workspace carve-up (256B aligned)
  char*  base = (char*)d_ws;
  size_t off  = 0;
  auto carve = [&](size_t bytes) {
    void* p = base + off;
    off = (off + bytes + 255) & ~(size_t)255;
    return p;
  };
  unsigned short* W1t  = (unsigned short*)carve((size_t)NP * KP1  * 2);
  unsigned short* W2t  = (unsigned short*)carve((size_t)NP * KP2  * 2);
  unsigned short* Wm1t = (unsigned short*)carve((size_t)NP * KPM1 * 2);
  unsigned short* Wm2t = (unsigned short*)carve((size_t)NP * KP2  * 2);
  unsigned short* a_bf = (unsigned short*)carve((size_t)MROWS * LDIM * 2);
  float*          lasts = (float*)carve((size_t)NDAG * LDIM * 4);

  // pack weights to padded, transposed bf16 (N x Kp)
  auto pk = [&](const float* s, unsigned short* d, int K, int N, int Kp, int Np) {
    int tot = Kp * Np;
    pack_bf16_t<<<(tot + 255) / 256, 256, 0, stream>>>(s, d, K, N, Kp, Np);
  };
  pk(W1,  W1t,  NATOM,  LDIM, KP1,  NP);
  pk(W2,  W2t,  LDIM,   LDIM, KP2,  NP);
  pk(Wm1, Wm1t, 2*LDIM, LDIM, KPM1, NP);
  pk(Wm2, Wm2t, LDIM,   LDIM, KP2,  NP);

  // stage 1: embedding GEMMs
  stage1_kernel<<<MROWS / MT, 256, 0, stream>>>(X, W1t, b1, W2t, b2, a_bf);

  // stage 2: recursive scan (16 DAGs per workgroup)
  stage2_kernel<<<NDAG / DAGS, 256, 0, stream>>>(a_bf, Wm1t, bm1, Wm2t, bm2, attn_w, lasts);

  // stage 3: pooling + classifier
  final_kernel<<<1, 256, 0, stream>>>(lasts, dag_w, Wf1, bf1, Wf2, bf2, Wf3, bf3, out);
}
